// EnergyMHA_26439818674788
// MI455X (gfx1250) — matrix operand order repaired and checked by reading, not compiled, not run on hardware
//
#include <hip/hip_runtime.h>

// ---------------------------------------------------------------------------
// EnergyMHA fwd+bwd for gfx1250 (MI455X), bf16 WMMA pipeline.
// B=8, L=1024, D=1024, H=16, HD=64.
// Batch is indexed by blockIdx.y; host runs all 8 batches concurrently if
// ws_size allows, else serializes with grid.y = 1.
// ---------------------------------------------------------------------------

typedef __bf16 bf16_t;
typedef bf16_t bf16x8 __attribute__((ext_vector_type(8)));
typedef bf16_t v16bf  __attribute__((ext_vector_type(16)));
typedef float  v8f    __attribute__((ext_vector_type(8)));

#define LSEQ   1024
#define DMODEL 1024
#define NHEAD  16
#define HDIM   64
#define NBATCH 8

// per-batch element strides
#define XBF_STRIDE   (1ull << 20)   // L*D
#define QK_STRIDE    (1ull << 20)   // H*L*HD
#define LSE_STRIDE   (1ull << 14)   // H*L
#define DQK_STRIDE   (1ull << 21)   // L*2048

union AFrag { v16bf v; bf16x8 h[2]; };
union CFrag { v8f v; float e[8]; };

static __device__ inline CFrag czero() {
  CFrag c;
#pragma unroll
  for (int i = 0; i < 8; ++i) c.e[i] = 0.f;
  return c;
}

// A-fragment (16x32 bf16): lane m = lane&15, ksel g = lane>>4.
// halves j: K = (j&7) + 8*g + 16*(j>>3)  -> two 16B loads per lane.
static __device__ inline v16bf load_a_frag(const bf16_t* base, int ld, int row0, int col0) {
  const int lane = threadIdx.x & 31;
  const int m = lane & 15, g = lane >> 4;
  const bf16_t* p = base + (size_t)(row0 + m) * ld + (col0 + 8 * g);
  AFrag f;
  f.h[0] = *(const bf16x8*)(p);
  f.h[1] = *(const bf16x8*)(p + 16);
  return f.v;
}

// B-fragment (32x16 bf16) loaded from row-major B^T (rows = N, cols = K):
// lane n = lane&15, halves j: K = (j&7) + 8*(lane>>4) + 16*(j>>3) -> two 16B loads per lane.
static __device__ inline v16bf load_b_frag(const bf16_t* base, int ld, int row0, int col0) {
  const int lane = threadIdx.x & 31;
  const int n = lane & 15, g = lane >> 4;
  const bf16_t* p = base + (size_t)(row0 + n) * ld + (col0 + 8 * g);
  AFrag f;
  f.h[0] = *(const bf16x8*)(p);
  f.h[1] = *(const bf16x8*)(p + 16);
  return f.v;
}

static __device__ inline v8f wmma_bf16(v16bf a, v16bf b, v8f c) {
  // (neg_a, A, neg_b, B, c_mod, C, reuse_a, reuse_b)
  return __builtin_amdgcn_wmma_f32_16x16x32_bf16(false, a, false, b, (short)0, c,
                                                 false, false);
}

// reductions within 16-lane halves (C-layout rows live per-half)
static __device__ inline float half_reduce_max(float x) {
#pragma unroll
  for (int m = 1; m < 16; m <<= 1) x = fmaxf(x, __shfl_xor(x, m, 32));
  return x;
}
static __device__ inline float half_reduce_sum(float x) {
#pragma unroll
  for (int m = 1; m < 16; m <<= 1) x += __shfl_xor(x, m, 32);
  return x;
}

// ---------------------------------------------------------------------------
// Weight conversion: Wcat[c][d] = (c<1024 ? Wq : Wk) row c, plus transpose.
__global__ void k_convert_w(const float* __restrict__ Wq, const float* __restrict__ Wk,
                            bf16_t* __restrict__ Wcat, bf16_t* __restrict__ WcatT) {
  int idx = blockIdx.x * 256 + threadIdx.x;   // 0 .. 2M-1
  int c = idx >> 10, d = idx & 1023;
  float v = (c < 1024) ? Wq[idx] : Wk[idx - (1 << 20)];
  bf16_t bv = (bf16_t)v;
  Wcat[idx] = bv;
  WcatT[d * 2048 + c] = bv;
}

__global__ void k_convert_x(const float* __restrict__ x, bf16_t* __restrict__ xbf) {
  const size_t boff = (size_t)blockIdx.y * XBF_STRIDE;
  size_t idx = (size_t)blockIdx.x * 256 + threadIdx.x;   // 0 .. 1M-1
  xbf[boff + idx] = (bf16_t)x[boff + idx];
}

// ---------------------------------------------------------------------------
// Projections: out[l, c] = sum_d xbf[l,d] * Wcat[c,d]   (out = Q|K, c in [0,2048))
// One wave computes a 32(M) x 64(N) tile (2 A-frags reused over 4 B-frags).
// Stores Q/K [h][l][z] and Qt/Kt [h][z][l].
__global__ void k_proj(const bf16_t* __restrict__ xbf, const bf16_t* __restrict__ Wcat,
                       bf16_t* __restrict__ Qm, bf16_t* __restrict__ Km,
                       bf16_t* __restrict__ Qt, bf16_t* __restrict__ Kt) {
  const size_t boff = (size_t)blockIdx.y * QK_STRIDE;
  const bf16_t* xb = xbf + (size_t)blockIdx.y * XBF_STRIDE;
  const int wave = blockIdx.x * 8 + (threadIdx.x >> 5);
  const int lane = threadIdx.x & 31;
  const int l0 = (wave >> 5) * 32;   // 32 row tiles of 32
  const int c0 = (wave & 31) * 64;   // 32 col strips of 64
  CFrag acc[2][4];
#pragma unroll
  for (int u = 0; u < 2; ++u)
#pragma unroll
    for (int t = 0; t < 4; ++t) acc[u][t] = czero();
  for (int kk = 0; kk < DMODEL; kk += 32) {
    v16bf a0 = load_a_frag(xb, DMODEL, l0, kk);
    v16bf a1 = load_a_frag(xb, DMODEL, l0 + 16, kk);
#pragma unroll
    for (int t = 0; t < 4; ++t) {
      v16bf bb = load_b_frag(Wcat, DMODEL, c0 + 16 * t, kk);
      acc[0][t].v = wmma_bf16(a0, bb, acc[0][t].v);
      acc[1][t].v = wmma_bf16(a1, bb, acc[1][t].v);
    }
  }
  const int n = lane & 15, mg = lane >> 4;
#pragma unroll
  for (int u = 0; u < 2; ++u) {
#pragma unroll
    for (int t = 0; t < 4; ++t) {
#pragma unroll
      for (int r = 0; r < 8; ++r) {
        int c = c0 + 16 * t + n;
        int l = l0 + 16 * u + 8 * mg + r;
        bf16_t bv = (bf16_t)acc[u][t].e[r];
        int h = (c & 1023) >> 6, z = c & 63;
        if (c < 1024) {
          Qm[boff + ((size_t)(h * LSEQ) + l) * HDIM + z] = bv;
          Qt[boff + ((size_t)(h * HDIM) + z) * LSEQ + l] = bv;
        } else {
          Km[boff + ((size_t)(h * LSEQ) + l) * HDIM + z] = bv;
          Kt[boff + ((size_t)(h * HDIM) + z) * LSEQ + l] = bv;
        }
      }
    }
  }
}

// ---------------------------------------------------------------------------
// Forward attention: per-wave 16 q rows of one head; online softmax over all k.
// Writes lse[h][q].
__global__ void k_attn_fwd(const bf16_t* __restrict__ Qm, const bf16_t* __restrict__ Km,
                           const float* __restrict__ beta, float* __restrict__ lse) {
  const size_t boff = (size_t)blockIdx.y * QK_STRIDE;
  float* lseb = lse + (size_t)blockIdx.y * LSE_STRIDE;
  const int wave = blockIdx.x * 8 + (threadIdx.x >> 5);
  const int lane = threadIdx.x & 31;
  const int h = wave >> 6;
  const int q0 = (wave & 63) * 16;
  const bf16_t* Qh = Qm + boff + (size_t)h * LSEQ * HDIM;
  const bf16_t* Kh = Km + boff + (size_t)h * LSEQ * HDIM;
  const float bta = beta[h];
  v16bf aq0 = load_a_frag(Qh, HDIM, q0, 0);
  v16bf aq1 = load_a_frag(Qh, HDIM, q0, 32);
  float mrow[8], srow[8];
#pragma unroll
  for (int r = 0; r < 8; ++r) { mrow[r] = -3.0e38f; srow[r] = 0.f; }
  for (int kc = 0; kc < LSEQ; kc += 64) {
    CFrag s[4];
#pragma unroll
    for (int t = 0; t < 4; ++t) {
      v16bf b0 = load_b_frag(Kh, HDIM, kc + 16 * t, 0);
      v16bf b1 = load_b_frag(Kh, HDIM, kc + 16 * t, 32);
      CFrag z = czero();
      z.v = wmma_bf16(aq0, b0, z.v);
      z.v = wmma_bf16(aq1, b1, z.v);
#pragma unroll
      for (int r = 0; r < 8; ++r) s[t].e[r] = z.e[r] * bta;
    }
#pragma unroll
    for (int r = 0; r < 8; ++r) {
      float cm = fmaxf(fmaxf(s[0].e[r], s[1].e[r]), fmaxf(s[2].e[r], s[3].e[r]));
      cm = half_reduce_max(cm);
      float nm = fmaxf(mrow[r], cm);
      float ps = __expf(s[0].e[r] - nm) + __expf(s[1].e[r] - nm) +
                 __expf(s[2].e[r] - nm) + __expf(s[3].e[r] - nm);
      ps = half_reduce_sum(ps);
      srow[r] = srow[r] * __expf(mrow[r] - nm) + ps;
      mrow[r] = nm;
    }
  }
  const int mg = lane >> 4;
  if ((lane & 15) == 0) {
#pragma unroll
    for (int r = 0; r < 8; ++r)
      lseb[h * LSEQ + q0 + 8 * mg + r] = mrow[r] + __logf(srow[r]);
  }
}

// y[b] = sum_h -(sum_q lse[h,q]) / beta[h]
__global__ void k_reduce_y(const float* __restrict__ lse, const float* __restrict__ beta,
                           float* __restrict__ yout) {
  __shared__ float sh[256];
  const float* lseb = lse + (size_t)blockIdx.y * LSE_STRIDE;
  float acc = 0.f;
  for (int i = threadIdx.x; i < NHEAD * LSEQ; i += 256)
    acc -= lseb[i] / beta[i >> 10];
  sh[threadIdx.x] = acc;
  __syncthreads();
  for (int s = 128; s > 0; s >>= 1) {
    if ((int)threadIdx.x < s) sh[threadIdx.x] += sh[threadIdx.x + s];
    __syncthreads();
  }
  if (threadIdx.x == 0) yout[blockIdx.y] = sh[0];
}

// ---------------------------------------------------------------------------
// dQ[q,h,z] = -sum_k P[q,k] K[k,z];  P = exp(beta*S - lse[q]).
// P tile is staged through LDS (C-layout -> A-fragment).
__global__ void k_bwd_dQ(const bf16_t* __restrict__ Qm, const bf16_t* __restrict__ Km,
                         const bf16_t* __restrict__ Kt, const float* __restrict__ beta,
                         const float* __restrict__ lse, bf16_t* __restrict__ dQK) {
  __shared__ bf16_t ptile[8][16 * 32];
  const size_t boff = (size_t)blockIdx.y * QK_STRIDE;
  const float* lseb = lse + (size_t)blockIdx.y * LSE_STRIDE;
  bf16_t* dQKb = dQK + (size_t)blockIdx.y * DQK_STRIDE;
  const int wid = threadIdx.x >> 5;
  const int lane = threadIdx.x & 31;
  const int wave = blockIdx.x * 8 + wid;
  const int h = wave >> 6;
  const int q0 = (wave & 63) * 16;
  const bf16_t* Qh = Qm + boff + (size_t)h * LSEQ * HDIM;
  const bf16_t* Kh = Km + boff + (size_t)h * LSEQ * HDIM;
  const bf16_t* Kth = Kt + boff + (size_t)h * HDIM * LSEQ;
  const float bta = beta[h];
  const int n = lane & 15, mg = lane >> 4;
  v16bf aq0 = load_a_frag(Qh, HDIM, q0, 0);
  v16bf aq1 = load_a_frag(Qh, HDIM, q0, 32);
  float lrow[8];
#pragma unroll
  for (int r = 0; r < 8; ++r) lrow[r] = lseb[h * LSEQ + q0 + 8 * mg + r];
  CFrag dacc[4];
#pragma unroll
  for (int t = 0; t < 4; ++t) dacc[t] = czero();
  bf16_t* myP = &ptile[wid][0];
  for (int kc = 0; kc < LSEQ; kc += 32) {
    CFrag s[2];
#pragma unroll
    for (int tt = 0; tt < 2; ++tt) {
      v16bf b0 = load_b_frag(Kh, HDIM, kc + 16 * tt, 0);
      v16bf b1 = load_b_frag(Kh, HDIM, kc + 16 * tt, 32);
      CFrag z = czero();
      z.v = wmma_bf16(aq0, b0, z.v);
      z.v = wmma_bf16(aq1, b1, z.v);
      s[tt] = z;
    }
    asm volatile("s_wait_dscnt 0" ::: "memory");  // prior aP loads done before overwrite
#pragma unroll
    for (int tt = 0; tt < 2; ++tt)
#pragma unroll
      for (int r = 0; r < 8; ++r) {
        float p = __expf(bta * s[tt].e[r] - lrow[r]);
        myP[(8 * mg + r) * 32 + (16 * tt + n)] = (bf16_t)p;
      }
    asm volatile("s_wait_dscnt 0" ::: "memory");  // stores visible before reload
    v16bf aP = load_a_frag(myP, 32, 0, 0);
#pragma unroll
    for (int t = 0; t < 4; ++t) {
      v16bf bk = load_b_frag(Kth, LSEQ, 16 * t, kc);
      dacc[t].v = wmma_bf16(aP, bk, dacc[t].v);
    }
  }
#pragma unroll
  for (int t = 0; t < 4; ++t)
#pragma unroll
    for (int r = 0; r < 8; ++r) {
      int l = q0 + 8 * mg + r;
      int c = h * HDIM + 16 * t + n;
      dQKb[(size_t)l * 2048 + c] = (bf16_t)(-dacc[t].e[r]);
    }
}

// dK[k,h,z] = -sum_q P[q,k] Q[q,z];  S^T tile = K Q^T, lse indexed along N (=q).
__global__ void k_bwd_dK(const bf16_t* __restrict__ Qm, const bf16_t* __restrict__ Km,
                         const bf16_t* __restrict__ Qt, const float* __restrict__ beta,
                         const float* __restrict__ lse, bf16_t* __restrict__ dQK) {
  __shared__ bf16_t ptile[8][16 * 32];
  const size_t boff = (size_t)blockIdx.y * QK_STRIDE;
  const float* lseb = lse + (size_t)blockIdx.y * LSE_STRIDE;
  bf16_t* dQKb = dQK + (size_t)blockIdx.y * DQK_STRIDE;
  const int wid = threadIdx.x >> 5;
  const int lane = threadIdx.x & 31;
  const int wave = blockIdx.x * 8 + wid;
  const int h = wave >> 6;
  const int k0 = (wave & 63) * 16;
  const bf16_t* Qh = Qm + boff + (size_t)h * LSEQ * HDIM;
  const bf16_t* Kh = Km + boff + (size_t)h * LSEQ * HDIM;
  const bf16_t* Qth = Qt + boff + (size_t)h * HDIM * LSEQ;
  const float bta = beta[h];
  const int n = lane & 15, mg = lane >> 4;
  v16bf ak0 = load_a_frag(Kh, HDIM, k0, 0);
  v16bf ak1 = load_a_frag(Kh, HDIM, k0, 32);
  CFrag dacc[4];
#pragma unroll
  for (int t = 0; t < 4; ++t) dacc[t] = czero();
  bf16_t* myP = &ptile[wid][0];
  for (int qc = 0; qc < LSEQ; qc += 32) {
    CFrag s[2];
    float lq[2];
#pragma unroll
    for (int tt = 0; tt < 2; ++tt) {
      v16bf b0 = load_b_frag(Qh, HDIM, qc + 16 * tt, 0);
      v16bf b1 = load_b_frag(Qh, HDIM, qc + 16 * tt, 32);
      CFrag z = czero();
      z.v = wmma_bf16(ak0, b0, z.v);
      z.v = wmma_bf16(ak1, b1, z.v);
      s[tt] = z;
      lq[tt] = lseb[h * LSEQ + qc + 16 * tt + n];
    }
    asm volatile("s_wait_dscnt 0" ::: "memory");
#pragma unroll
    for (int tt = 0; tt < 2; ++tt)
#pragma unroll
      for (int r = 0; r < 8; ++r) {
        float p = __expf(bta * s[tt].e[r] - lq[tt]);
        myP[(8 * mg + r) * 32 + (16 * tt + n)] = (bf16_t)p;
      }
    asm volatile("s_wait_dscnt 0" ::: "memory");
    v16bf aP = load_a_frag(myP, 32, 0, 0);
#pragma unroll
    for (int t = 0; t < 4; ++t) {
      v16bf bq = load_b_frag(Qth, LSEQ, 16 * t, qc);
      dacc[t].v = wmma_bf16(aP, bq, dacc[t].v);
    }
  }
#pragma unroll
  for (int t = 0; t < 4; ++t)
#pragma unroll
    for (int r = 0; r < 8; ++r) {
      int l = k0 + 8 * mg + r;
      int c = 1024 + h * HDIM + 16 * t + n;
      dQKb[(size_t)l * 2048 + c] = (bf16_t)(-dacc[t].e[r]);
    }
}

// ---------------------------------------------------------------------------
// grads[l, d] = sum_c dQK[l, c] * Wcat[c, d]   (K = 2048), f32 output.
// One wave computes a 32(M) x 64(N) tile.
__global__ void k_grad(const bf16_t* __restrict__ dQK, const bf16_t* __restrict__ WcatT,
                       float* __restrict__ grads) {
  const bf16_t* dQKb = dQK + (size_t)blockIdx.y * DQK_STRIDE;
  float* gb = grads + (size_t)blockIdx.y * XBF_STRIDE;
  const int wave = blockIdx.x * 8 + (threadIdx.x >> 5);
  const int lane = threadIdx.x & 31;
  const int l0 = (wave >> 4) * 32;   // 32 row tiles of 32
  const int d0 = (wave & 15) * 64;   // 16 col strips of 64
  CFrag acc[2][4];
#pragma unroll
  for (int u = 0; u < 2; ++u)
#pragma unroll
    for (int t = 0; t < 4; ++t) acc[u][t] = czero();
  for (int c = 0; c < 2048; c += 32) {
    v16bf a0 = load_a_frag(dQKb, 2048, l0, c);
    v16bf a1 = load_a_frag(dQKb, 2048, l0 + 16, c);
#pragma unroll
    for (int t = 0; t < 4; ++t) {
      v16bf bb = load_b_frag(WcatT, 2048, d0 + 16 * t, c);
      acc[0][t].v = wmma_bf16(a0, bb, acc[0][t].v);
      acc[1][t].v = wmma_bf16(a1, bb, acc[1][t].v);
    }
  }
  const int n = lane & 15, mg = lane >> 4;
#pragma unroll
  for (int u = 0; u < 2; ++u)
#pragma unroll
    for (int t = 0; t < 4; ++t)
#pragma unroll
      for (int r = 0; r < 8; ++r)
        gb[(size_t)(l0 + 16 * u + 8 * mg + r) * DMODEL + d0 + 16 * t + n] =
            acc[u][t].e[r];
}

// ---------------------------------------------------------------------------
extern "C" void kernel_launch(void* const* d_in, const int* in_sizes, int n_in,
                              void* d_out, int out_size, void* d_ws, size_t ws_size,
                              hipStream_t stream) {
  const float* x    = (const float*)d_in[0];
  const float* Wq   = (const float*)d_in[1];
  const float* Wk   = (const float*)d_in[2];
  const float* beta = (const float*)d_in[3];
  float* yout = (float*)d_out;
  float* gout = (float*)d_out + NBATCH;

  // Workspace layout: batched (nb=8) if it fits, else single-batch reuse (nb=1).
  const size_t per_batch = (2ull << 20)        // xbf
                         + 4ull * (2ull << 20) // Qm, Km, Qt, Kt
                         + (64ull << 10)       // lse
                         + (4ull << 20);       // dQK
  const size_t shared_w = 8ull << 20;          // Wcat + WcatT
  const int nb = (ws_size >= shared_w + (size_t)NBATCH * per_batch) ? NBATCH : 1;

  char* p = (char*)d_ws;
  bf16_t* xbf   = (bf16_t*)p; p += (size_t)nb * (2ull << 20);
  bf16_t* Wcat  = (bf16_t*)p; p += 4ull << 20;
  bf16_t* WcatT = (bf16_t*)p; p += 4ull << 20;
  bf16_t* Qm    = (bf16_t*)p; p += (size_t)nb * (2ull << 20);
  bf16_t* Km    = (bf16_t*)p; p += (size_t)nb * (2ull << 20);
  bf16_t* Qt    = (bf16_t*)p; p += (size_t)nb * (2ull << 20);
  bf16_t* Kt    = (bf16_t*)p; p += (size_t)nb * (2ull << 20);
  float*  lse   = (float*)p;  p += (size_t)nb * (64ull << 10);
  bf16_t* dQK   = (bf16_t*)p;

  k_convert_w<<<dim3(8192, 1), 256, 0, stream>>>(Wq, Wk, Wcat, WcatT);

  const int outer = NBATCH / nb;
  for (int i = 0; i < outer; ++i) {
    const float* xb = x + (size_t)i * nb * XBF_STRIDE;
    float* gb = gout + (size_t)i * nb * XBF_STRIDE;
    float* yb = yout + (size_t)i * nb;
    k_convert_x<<<dim3(4096, nb), 256, 0, stream>>>(xb, xbf);
    k_proj<<<dim3(128, nb), 256, 0, stream>>>(xbf, Wcat, Qm, Km, Qt, Kt);
    k_attn_fwd<<<dim3(128, nb), 256, 0, stream>>>(Qm, Km, beta, lse);
    k_reduce_y<<<dim3(1, nb), 256, 0, stream>>>(lse, beta, yb);
    k_bwd_dQ<<<dim3(128, nb), 256, 0, stream>>>(Qm, Km, Kt, beta, lse, dQK);
    k_bwd_dK<<<dim3(128, nb), 256, 0, stream>>>(Qm, Km, Qt, beta, lse, dQK);
    k_grad<<<dim3(64, nb), 256, 0, stream>>>(dQK, WcatT, gb);
  }
  (void)in_sizes; (void)n_in; (void)out_size;
}
